// MultiHeadAttentionWithPE_67010079752321
// MI455X (gfx1250) — matrix operand order repaired and checked by reading, not compiled, not run on hardware
//
#include <hip/hip_runtime.h>
#include <hip/hip_bf16.h>

typedef __attribute__((ext_vector_type(16))) _Float16 v16h;
typedef __attribute__((ext_vector_type(8)))  float    v8f;
typedef __attribute__((ext_vector_type(4)))  int      i4;

#define D_MODEL 768
#define N_HEADS 12
#define HEAD_DIM 64
#define BB 2
#define TT 2048
#define BT (BB * TT)      // 4096 rows
#define LDS_STRIDE 48     // halves per row in LDS staging (96B, 16B aligned)

union Frag16 { i4 q[2]; v16h v; };

__device__ inline v8f zero8() {
  v8f z;
  #pragma unroll
  for (int i = 0; i < 8; ++i) z[i] = 0.0f;
  return z;
}

// 32-bit LDS byte offset for the async-to-LDS VDST operand.
__device__ inline unsigned ldsoff(const void* p) {
  return (unsigned)(size_t)(__attribute__((address_space(3))) const void*)p;
}

// ---------------------------------------------------------------------------
// One-time f32 -> f16 conversion (x, and attn path stays f16 end-to-end).
// ---------------------------------------------------------------------------
__global__ __launch_bounds__(256)
void cvt_f32_f16(const float* __restrict__ src, _Float16* __restrict__ dst, int n) {
  const int i = blockIdx.x * 256 + threadIdx.x;
  if (i < n) dst[i] = (_Float16)src[i];
}

// One-time weight pack: WT[n][k] = (f16) W[k][n]  (768x768)
__global__ __launch_bounds__(256)
void pack_wt(const float* __restrict__ W, _Float16* __restrict__ WT) {
  const int i = blockIdx.x * 256 + threadIdx.x;
  if (i >= D_MODEL * D_MODEL) return;
  const int n = i % D_MODEL, k = i / D_MODEL;   // read coalesced along n
  WT[(size_t)n * D_MODEL + k] = (_Float16)W[(size_t)k * D_MODEL + n];
}

// ---------------------------------------------------------------------------
// C[BT x 768] (f32) = Xh[BT x 768] (f16) * WT^T, WT given as [n][k] f16.
// Block = 128 threads (4 waves). Block tile 64x64, K step 32.
// Tiles staged via CDNA5 async global->LDS copies (ASYNCcnt).
// ---------------------------------------------------------------------------
__global__ __launch_bounds__(128)
void gemm768_wmma(const _Float16* __restrict__ Xh, const _Float16* __restrict__ WT,
                  float* __restrict__ C) {
  __shared__ _Float16 Xs[64 * LDS_STRIDE];   // [row][k]
  __shared__ _Float16 Ws[64 * LDS_STRIDE];   // [n][k]

  const int bid   = blockIdx.x;
  const int mtile = bid % (BT / 64);
  const int ntile = bid / (BT / 64);
  const int mbase = mtile * 64, nbase = ntile * 64;
  const int tid  = threadIdx.x;
  const int lane = tid & 31, wave = tid >> 5;
  const int g = lane >> 4, lm = lane & 15;

  // staging assignment: thread copies 32 bytes (16 halves) of one tile row
  const int r  = tid >> 1;
  const int kk = (tid & 1) * 16;
  const unsigned ldsX = ldsoff(&Xs[r * LDS_STRIDE + kk]);
  const unsigned ldsW = ldsoff(&Ws[r * LDS_STRIDE + kk]);
  const _Float16* gxp = Xh + (size_t)(mbase + r) * D_MODEL + kk;
  const _Float16* gwp = WT + (size_t)(nbase + r) * D_MODEL + kk;

  v8f acc[4];
  #pragma unroll
  for (int i = 0; i < 4; ++i) acc[i] = zero8();

  for (int kb = 0; kb < D_MODEL; kb += 32) {
    const unsigned long long gx = (unsigned long long)(gxp + kb);
    const unsigned long long gw = (unsigned long long)(gwp + kb);
    asm volatile("global_load_async_to_lds_b128 %0, %1, off"
                 :: "v"(ldsX), "v"(gx) : "memory");
    asm volatile("global_load_async_to_lds_b128 %0, %1, off offset:16"
                 :: "v"(ldsX), "v"(gx) : "memory");
    asm volatile("global_load_async_to_lds_b128 %0, %1, off"
                 :: "v"(ldsW), "v"(gw) : "memory");
    asm volatile("global_load_async_to_lds_b128 %0, %1, off offset:16"
                 :: "v"(ldsW), "v"(gw) : "memory");
    if (kb + 32 < D_MODEL) {
      __builtin_prefetch(gxp + kb + 32, 0, 1);
      __builtin_prefetch(gwp + kb + 32, 0, 1);
    }
    asm volatile("s_wait_asynccnt 0" ::: "memory");
    __syncthreads();

    // A fragment: 16x32, lane holds row (wave*16+lm); chunks at k = 8g and 16+8g
    Frag16 a;
    const _Float16* ar = &Xs[(wave * 16 + lm) * LDS_STRIDE];
    a.q[0] = *(const i4*)(ar + g * 8);
    a.q[1] = *(const i4*)(ar + 16 + g * 8);

    #pragma unroll
    for (int nc = 0; nc < 4; ++nc) {
      // B fragment: lane holds column (nc*16+lm); chunks at k = 8g and 16+8g
      Frag16 bf;
      const _Float16* br = &Ws[(nc * 16 + lm) * LDS_STRIDE + g * 8];
      bf.q[0] = ((const i4*)br)[0];
      bf.q[1] = ((const i4*)br)[2];
      acc[nc] = __builtin_amdgcn_wmma_f32_16x16x32_f16(
          false, a.v, false, bf.v, (short)0, acc[nc], false, false);
    }
    __syncthreads();
  }

  // Epilogue: C layout — lane holds (m = v + 8g, n = lm) per tile
  #pragma unroll
  for (int nc = 0; nc < 4; ++nc)
    #pragma unroll
    for (int v = 0; v < 8; ++v) {
      const int m = mbase + wave * 16 + v + 8 * g;
      const int n = nbase + nc * 16 + lm;
      C[(size_t)m * D_MODEL + n] = acc[nc][v];
    }
}

// ---------------------------------------------------------------------------
// RoPE on Q,K + repack to f16 [B,H,T,64]; V -> f16 transposed [B,H,64,T]
// ---------------------------------------------------------------------------
__global__ __launch_bounds__(256)
void rope_pack(const float* __restrict__ Qf, const float* __restrict__ Kf,
               const float* __restrict__ Vf, _Float16* __restrict__ Qh,
               _Float16* __restrict__ Kh, _Float16* __restrict__ Vt) {
  const int idx = blockIdx.x * 256 + threadIdx.x;
  if (idx >= BT * D_MODEL) return;
  const int d768 = idx % D_MODEL;
  const int t    = (idx / D_MODEL) % TT;
  const int b    = idx / (D_MODEL * TT);
  const int h = d768 >> 6, d = d768 & 63;
  const int i = d & 31;

  // inv_freq = 10000^(-i/32) = exp(-i * ln(10000)/32)
  const float inv_freq = __expf(-(float)i * (9.210340371976184f / 32.0f));
  const float ang = (float)t * inv_freq;
  float s, c;
  __sincosf(ang, &s, &c);

  const size_t base = (size_t)(b * TT + t) * D_MODEL + h * 64;
  const float q1 = Qf[base + i], q2 = Qf[base + i + 32];
  const float k1 = Kf[base + i], k2 = Kf[base + i + 32];
  const float qo = (d < 32) ? (q1 * c - q2 * s) : (q1 * s + q2 * c);
  const float ko = (d < 32) ? (k1 * c - k2 * s) : (k1 * s + k2 * c);

  const size_t rb = ((size_t)(b * N_HEADS + h) * TT + t) * 64 + d;
  Qh[rb] = (_Float16)qo;
  Kh[rb] = (_Float16)ko;
  Vt[((size_t)(b * N_HEADS + h) * 64 + d) * TT + t] = (_Float16)Vf[base + d];
}

// ---------------------------------------------------------------------------
// Flash-style causal attention. Block = 128 threads (4 independent waves).
// Wave owns 16 query rows; 32 keys per iteration; online softmax.
// Output attn [B,T,768] f16 (feeds final WMMA projection directly).
// ---------------------------------------------------------------------------
__global__ __launch_bounds__(128)
void flash_attn(const _Float16* __restrict__ Qh, const _Float16* __restrict__ Kh,
                const _Float16* __restrict__ Vt, _Float16* __restrict__ O) {
  __shared__ _Float16 Pl[4][16 * LDS_STRIDE];  // per-wave P staging

  const int bid = blockIdx.x;
  const int qt  = bid % (TT / 64);
  const int h   = (bid / (TT / 64)) % N_HEADS;
  const int b   = bid / ((TT / 64) * N_HEADS);
  const int tid = threadIdx.x, lane = tid & 31, wave = tid >> 5;
  const int g = lane >> 4, lm = lane & 15;
  const int qw = qt * 64 + wave * 16;          // this wave's first query row

  const _Float16* Qp = Qh + (size_t)(b * N_HEADS + h) * TT * 64;
  const _Float16* Kp = Kh + (size_t)(b * N_HEADS + h) * TT * 64;
  const _Float16* Vp = Vt + (size_t)(b * N_HEADS + h) * 64 * TT;
  _Float16* Pw = &Pl[wave][0];

  // Q A-fragments for rows qw+lm (two 16x32 frags covering HEAD_DIM=64)
  Frag16 qa[2];
  const _Float16* qrow = Qp + (size_t)(qw + lm) * 64;
  #pragma unroll
  for (int f = 0; f < 2; ++f) {
    qa[f].q[0] = *(const i4*)(qrow + f * 32 + g * 8);
    qa[f].q[1] = *(const i4*)(qrow + f * 32 + 16 + g * 8);
  }

  v8f oacc[4];
  #pragma unroll
  for (int i = 0; i < 4; ++i) oacc[i] = zero8();
  float mrow[8], lrow[8];
  #pragma unroll
  for (int v = 0; v < 8; ++v) { mrow[v] = -1.0e30f; lrow[v] = 0.0f; }

  const int kend = qw + 16;                    // causal: keys 0..qw+15
  for (int kb = 0; kb < kend; kb += 32) {
    // ---- S = Q * K^T for 32 keys: two 16x16 tiles ----
    v8f c0 = zero8(), c1 = zero8();
    #pragma unroll
    for (int f = 0; f < 2; ++f) {
      Frag16 b0, b1;
      const _Float16* kr0 = Kp + (size_t)(kb + lm) * 64 + f * 32 + g * 8;
      b0.q[0] = ((const i4*)kr0)[0];
      b0.q[1] = ((const i4*)kr0)[2];
      const _Float16* kr1 = Kp + (size_t)(kb + 16 + lm) * 64 + f * 32 + g * 8;
      b1.q[0] = ((const i4*)kr1)[0];
      b1.q[1] = ((const i4*)kr1)[2];
      c0 = __builtin_amdgcn_wmma_f32_16x16x32_f16(false, qa[f].v, false, b0.v,
                                                  (short)0, c0, false, false);
      c1 = __builtin_amdgcn_wmma_f32_16x16x32_f16(false, qa[f].v, false, b1.v,
                                                  (short)0, c1, false, false);
    }

    // ---- scale, causal mask, online softmax ----
    #pragma unroll
    for (int v = 0; v < 8; ++v) {
      const int q = qw + v + 8 * g;
      float s0 = c0[v] * 0.125f;               // 1/sqrt(64)
      float s1 = c1[v] * 0.125f;
      if (kb + lm      > q) s0 = -1.0e30f;
      if (kb + 16 + lm > q) s1 = -1.0e30f;

      float mx = fmaxf(s0, s1);
      #pragma unroll
      for (int off = 8; off >= 1; off >>= 1)
        mx = fmaxf(mx, __shfl_xor(mx, off));
      const float mn = fmaxf(mrow[v], mx);
      const float alpha = __expf(mrow[v] - mn);
      const float p0 = __expf(s0 - mn);
      const float p1 = __expf(s1 - mn);
      float rs = p0 + p1;
      #pragma unroll
      for (int off = 8; off >= 1; off >>= 1)
        rs += __shfl_xor(rs, off);
      lrow[v] = lrow[v] * alpha + rs;
      mrow[v] = mn;
      #pragma unroll
      for (int dc = 0; dc < 4; ++dc) oacc[dc][v] *= alpha;

      // stash P (f16) for the layout transpose: [row][key]
      Pw[(v + 8 * g) * LDS_STRIDE + lm]      = (_Float16)p0;
      Pw[(v + 8 * g) * LDS_STRIDE + 16 + lm] = (_Float16)p1;
    }
    // per-wave LDS RAW across lanes: same-wave DS ops are in-order; just drain
    asm volatile("s_wait_dscnt 0" ::: "memory");

    // ---- O += P * V (P: 16x32 A-frag from LDS; V^T rows give B-frags) ----
    Frag16 pa;
    const _Float16* pr = &Pw[lm * LDS_STRIDE];
    pa.q[0] = *(const i4*)(pr + g * 8);
    pa.q[1] = *(const i4*)(pr + 16 + g * 8);
    #pragma unroll
    for (int dc = 0; dc < 4; ++dc) {
      Frag16 bv;
      const _Float16* vr = Vp + (size_t)(dc * 16 + lm) * TT + kb + g * 8;
      bv.q[0] = ((const i4*)vr)[0];
      bv.q[1] = ((const i4*)vr)[2];
      oacc[dc] = __builtin_amdgcn_wmma_f32_16x16x32_f16(
          false, pa.v, false, bv.v, (short)0, oacc[dc], false, false);
    }
    asm volatile("s_wait_dscnt 0" ::: "memory");  // protect Pw before next iter's stores
  }

  // ---- epilogue: O /= l, write [B,T,H*64] f16 ----
  #pragma unroll
  for (int dc = 0; dc < 4; ++dc)
    #pragma unroll
    for (int v = 0; v < 8; ++v) {
      const int q = qw + v + 8 * g;
      O[(size_t)(b * TT + q) * D_MODEL + h * 64 + dc * 16 + lm] =
          (_Float16)(oacc[dc][v] / lrow[v]);
    }
}

// ---------------------------------------------------------------------------
extern "C" void kernel_launch(void* const* d_in, const int* in_sizes, int n_in,
                              void* d_out, int out_size, void* d_ws, size_t ws_size,
                              hipStream_t stream) {
  (void)in_sizes; (void)n_in; (void)out_size; (void)ws_size;
  const float* x  = (const float*)d_in[0];
  const float* Wq = (const float*)d_in[1];
  const float* Wk = (const float*)d_in[2];
  const float* Wv = (const float*)d_in[3];
  const float* Wo = (const float*)d_in[4];
  float* out = (float*)d_out;

  char* ws = (char*)d_ws;
  size_t off = 0;
  auto alloc = [&](size_t bytes) -> char* {
    char* p = ws + off;
    off += (bytes + 255) & ~(size_t)255;
    return p;
  };
  const size_t nElem = (size_t)BT * D_MODEL;         // 3,145,728
  const size_t wElem = (size_t)D_MODEL * D_MODEL;    //   589,824
  float*    Qf    = (float*)alloc(nElem * 4);
  float*    Kf    = (float*)alloc(nElem * 4);
  float*    Vf    = (float*)alloc(nElem * 4);
  _Float16* xh    = (_Float16*)alloc(nElem * 2);
  _Float16* attnH = (_Float16*)alloc(nElem * 2);
  _Float16* Qh    = (_Float16*)alloc(nElem * 2);
  _Float16* Kh    = (_Float16*)alloc(nElem * 2);
  _Float16* Vt    = (_Float16*)alloc(nElem * 2);
  _Float16* WTq   = (_Float16*)alloc(wElem * 2);
  _Float16* WTk   = (_Float16*)alloc(wElem * 2);
  _Float16* WTv   = (_Float16*)alloc(wElem * 2);
  _Float16* WTo   = (_Float16*)alloc(wElem * 2);

  // one-time precision/layout packing (all resident in L2 afterwards)
  cvt_f32_f16<<<(int)((nElem + 255) / 256), 256, 0, stream>>>(x, xh, (int)nElem);
  const int wBlocks = (int)((wElem + 255) / 256);
  pack_wt<<<wBlocks, 256, 0, stream>>>(Wq, WTq);
  pack_wt<<<wBlocks, 256, 0, stream>>>(Wk, WTk);
  pack_wt<<<wBlocks, 256, 0, stream>>>(Wv, WTv);
  pack_wt<<<wBlocks, 256, 0, stream>>>(Wo, WTo);

  const int gemmBlocks = (BT / 64) * (D_MODEL / 64);   // 768
  gemm768_wmma<<<gemmBlocks, 128, 0, stream>>>(xh, WTq, Qf);
  gemm768_wmma<<<gemmBlocks, 128, 0, stream>>>(xh, WTk, Kf);
  gemm768_wmma<<<gemmBlocks, 128, 0, stream>>>(xh, WTv, Vf);

  rope_pack<<<(int)((nElem + 255) / 256), 256, 0, stream>>>(Qf, Kf, Vf, Qh, Kh, Vt);

  const int flashBlocks = BB * N_HEADS * (TT / 64);    // 768
  flash_attn<<<flashBlocks, 128, 0, stream>>>(Qh, Kh, Vt, attnH);

  gemm768_wmma<<<gemmBlocks, 128, 0, stream>>>(attnH, WTo, out);
}
